// CrossModalAttention_80049600463725
// MI455X (gfx1250) — matrix operand order repaired and checked by reading, compiled, not run on hardware
//
#include <hip/hip_runtime.h>
#include <hip/hip_bf16.h>
#include <cstdint>
#include <cstddef>

// ---------------------------------------------------------------------------
// CDNA5 (gfx1250, wave32) cross-modal attention, all matmuls via
// v_wmma_f32_16x16x32_bf16.  fp32 inputs are converted to bf16 once; all
// accumulation is fp32.  Softmax uses the clamp(+/-1e4) + online-max scheme
// from the reference.
// ---------------------------------------------------------------------------

typedef __attribute__((ext_vector_type(16))) __bf16 v16bf;
typedef __attribute__((ext_vector_type(8)))  __bf16 v8bf;
typedef __attribute__((ext_vector_type(8)))  float  v8f;

union BF16Frag { v16bf v; v8bf h[2]; };

__device__ __forceinline__ __bf16 f2bf(float f) {
  union { float f; uint32_t u; } in; in.f = f;
  uint32_t u = in.u;
  uint32_t r = u + 0x7FFFu + ((u >> 16) & 1u);   // round-to-nearest-even
  union { uint16_t s; __bf16 b; } out; out.s = (uint16_t)(r >> 16);
  return out.b;
}

// ------------------------- elementwise conversions -------------------------

__global__ void cvt_f32_bf16(const float* __restrict__ src,
                             __bf16* __restrict__ dst, int n) {
  int i = blockIdx.x * blockDim.x + threadIdx.x;
  if (i < n) dst[i] = f2bf(src[i]);
}

// W is [K][N] row-major fp32 -> WT is [N][K] row-major bf16
__global__ void xpose_f32_bf16(const float* __restrict__ W,
                               __bf16* __restrict__ WT, int K, int N) {
  int i = blockIdx.x * blockDim.x + threadIdx.x;
  if (i >= K * N) return;
  int n = i / K;
  int k = i % K;
  WT[i] = f2bf(W[(size_t)k * N + n]);
}

// ------------------------------- WMMA GEMM ---------------------------------
// C[M,N] = A[M,K] @ BT[N,K]^T + bias, stored per `mode`:
//   mode 1: bf16 head-major      out[((b*Hn+h)*seqLen + n)*64 + d]
//   mode 2: fp32 flat            out[R*N + C]
//   mode 3: bf16 head-transposed out[((b*Hn+h)*64 + d)*seqLen + n]
// Block = 256 threads = 8 waves (4x2); wave tile 32x64 (2x4 accumulators,
// B fragments shared across the two row tiles: 8 WMMAs per 12 b128 loads),
// block tile 128x128.  K must be a multiple of 64 (holds: 1024 / 768).
// Two-stage software pipeline: two fragment buffer sets alternate per
// 32-deep K step so WMMAs overlap the next step's global_load_b128s.

__global__ __launch_bounds__(256) void gemm_bf16_wmma(
    const __bf16* __restrict__ A, const __bf16* __restrict__ BT,
    const float* __restrict__ bias, void* __restrict__ Out,
    int M, int N, int K, int mode, int seqLen, int Hn)
{
  const int lane = threadIdx.x & 31;
  const int wave = threadIdx.x >> 5;
  const int l  = lane & 15;
  const int hl = lane >> 4;                 // wave32 half
  const int wr = wave >> 1, wc = wave & 1;
  const int row0 = blockIdx.y * 128 + wr * 32;
  const int col0 = blockIdx.x * 128 + wc * 64;

  const __bf16* arow0 = A  + (size_t)(row0 + l) * K;        // A frag rows 0..15
  const __bf16* arow1 = A  + (size_t)(row0 + 16 + l) * K;   // A frag rows 16..31
  const __bf16* brow  = BT + (size_t)(col0 + l) * K;        // B frag: N = lane&15

  v8f acc[2][4];
#pragma unroll
  for (int i = 0; i < 2; ++i)
#pragma unroll
    for (int c = 0; c < 4; ++c)
      acc[i][c] = (v8f){0.f,0.f,0.f,0.f,0.f,0.f,0.f,0.f};

  // fragment loader for one 32-deep K step
  auto loadStep = [&](BF16Frag* af, BF16Frag* bf, int k0) {
    // A layout: half0 K runs [0..7],[16..23]; half1 [8..15],[24..31]
    af[0].h[0] = *(const v8bf*)(arow0 + k0 + hl * 8);
    af[0].h[1] = *(const v8bf*)(arow0 + k0 + 16 + hl * 8);
    af[1].h[0] = *(const v8bf*)(arow1 + k0 + hl * 8);
    af[1].h[1] = *(const v8bf*)(arow1 + k0 + 16 + hl * 8);
    // B layout: same as A: half0 K runs [0..7],[16..23]; half1 [8..15],[24..31]
#pragma unroll
    for (int c = 0; c < 4; ++c) {
      const __bf16* bp = brow + (size_t)c * 16 * K + k0 + hl * 8;
      bf[c].h[0] = *(const v8bf*)bp;
      bf[c].h[1] = *(const v8bf*)(bp + 16);
    }
  };
  auto wmma8 = [&](BF16Frag* af, BF16Frag* bf) {
#pragma unroll
    for (int i = 0; i < 2; ++i)
#pragma unroll
      for (int c = 0; c < 4; ++c)
        acc[i][c] = __builtin_amdgcn_wmma_f32_16x16x32_bf16(
            false, af[i].v, false, bf[c].v, (short)0, acc[i][c], false, false);
  };

  BF16Frag afA[2], bfA[4], afB[2], bfB[4];
  loadStep(afA, bfA, 0);
  for (int k0 = 0; k0 < K; k0 += 64) {
    loadStep(afB, bfB, k0 + 32);            // stage B for second half-step
    if (k0 + 96 < K) {
      __builtin_prefetch(arow0 + k0 + 128, 0, 3);
      __builtin_prefetch(arow1 + k0 + 128, 0, 3);
      __builtin_prefetch(brow + k0 + 128, 0, 3);
    }
    wmma8(afA, bfA);                        // compute first half-step
    if (k0 + 64 < K) loadStep(afA, bfA, k0 + 64);  // stage A for next iter
    wmma8(afB, bfB);                        // compute second half-step
  }

#pragma unroll
  for (int i = 0; i < 2; ++i)
#pragma unroll
    for (int c = 0; c < 4; ++c) {
      const int C = col0 + c * 16 + l;
      const float bb = bias[C];
#pragma unroll
      for (int r = 0; r < 8; ++r) {
        const int R = row0 + i * 16 + hl * 8 + r;  // C layout: row = vgpr+8*half
        float val = acc[i][c][r] + bb;
        if (mode == 2) {
          ((float*)Out)[(size_t)R * N + C] = val;
        } else {
          const int b = R / seqLen, n = R % seqLen;
          const int h = C >> 6, d = C & 63;
          __bf16* o = (__bf16*)Out;
          if (mode == 1)
            o[(((size_t)(b * Hn + h) * seqLen + n) << 6) + d] = f2bf(val);
          else
            o[((size_t)(b * Hn + h) * 64 + d) * seqLen + n] = f2bf(val);
        }
      }
    }
}

// ---------------------------- flash attention ------------------------------
// Qh: [B*H, Nq, 64] bf16, Kh: [B*H, Nkv, 64] bf16, Vt: [B*H, 64, Nkv] bf16.
// AO: [B*Nq, H*64] bf16 (attention output, input of the O projection).
// 4 waves/block, each wave owns one 16-row q tile; KV walked in 64-column
// chunks (4 S tiles per online-softmax pass to amortize the cross-lane
// reductions and the O rescale).

__global__ __launch_bounds__(128) void flash_attn_bf16(
    const __bf16* __restrict__ Qh, const __bf16* __restrict__ Kh,
    const __bf16* __restrict__ Vt, __bf16* __restrict__ AO,
    int Nq, int Nkv, int Hn)
{
  __shared__ __bf16 ldsP[4 * 16 * 64];      // per-wave 16x64 P tile (2 KB)

  const int tid  = threadIdx.x;
  const int wave = tid >> 5;
  const int lane = tid & 31;
  const int l  = lane & 15;
  const int hl = lane >> 4;
  const int bh = blockIdx.y;                // b*H + h
  const int q0 = (blockIdx.x * 4 + wave) * 16;

  const __bf16* Qbase = Qh + (size_t)bh * Nq  * 64;
  const __bf16* Kbase = Kh + (size_t)bh * Nkv * 64;
  const __bf16* Vbase = Vt + (size_t)bh * 64 * (size_t)Nkv;
  __bf16* pbuf = ldsP + wave * (16 * 64);

  // Q fragments (A-matrix, K-dim = head dim, two 32-deep steps)
  BF16Frag aq[2];
  {
    const __bf16* qrow = Qbase + (size_t)(q0 + l) * 64;
#pragma unroll
    for (int t = 0; t < 2; ++t) {
      aq[t].h[0] = *(const v8bf*)(qrow + t * 32 + hl * 8);
      aq[t].h[1] = *(const v8bf*)(qrow + t * 32 + 16 + hl * 8);
    }
  }

  float mrow[8], lrow[8];
  v8f o[4];
#pragma unroll
  for (int r = 0; r < 8; ++r) { mrow[r] = -3.0e38f; lrow[r] = 0.f; }
#pragma unroll
  for (int dt = 0; dt < 4; ++dt)
    o[dt] = (v8f){0.f,0.f,0.f,0.f,0.f,0.f,0.f,0.f};

  for (int j0 = 0; j0 < Nkv; j0 += 64) {
    // ---- S = Q @ K^T for 64 kv columns (four 16x16 tiles) ----
    v8f s[4];
#pragma unroll
    for (int c = 0; c < 4; ++c)
      s[c] = (v8f){0.f,0.f,0.f,0.f,0.f,0.f,0.f,0.f};
#pragma unroll
    for (int t = 0; t < 2; ++t) {
#pragma unroll
      for (int c = 0; c < 4; ++c) {
        const __bf16* kr = Kbase + (size_t)(j0 + c * 16 + l) * 64;
        BF16Frag bk;
        bk.h[0] = *(const v8bf*)(kr + t * 32 + hl * 8);
        bk.h[1] = *(const v8bf*)(kr + t * 32 + 16 + hl * 8);
        s[c] = __builtin_amdgcn_wmma_f32_16x16x32_bf16(
            false, aq[t].v, false, bk.v, (short)0, s[c], false, false);
      }
    }

    // ---- scale, clamp, online softmax over the 64-column chunk ----
    float fexp[8];
#pragma unroll
    for (int r = 0; r < 8; ++r) {
      float a0 = fminf(fmaxf(s[0][r] * 0.125f, -1.0e4f), 1.0e4f);
      float a1 = fminf(fmaxf(s[1][r] * 0.125f, -1.0e4f), 1.0e4f);
      float a2 = fminf(fmaxf(s[2][r] * 0.125f, -1.0e4f), 1.0e4f);
      float a3 = fminf(fmaxf(s[3][r] * 0.125f, -1.0e4f), 1.0e4f);
      float rm = fmaxf(fmaxf(a0, a1), fmaxf(a2, a3));
#pragma unroll
      for (int off = 1; off < 16; off <<= 1)
        rm = fmaxf(rm, __shfl_xor(rm, off, 16));
      float mn = fmaxf(mrow[r], rm);
      float f  = __expf(mrow[r] - mn);
      float p0 = __expf(a0 - mn);
      float p1 = __expf(a1 - mn);
      float p2 = __expf(a2 - mn);
      float p3 = __expf(a3 - mn);
      float rs = (p0 + p1) + (p2 + p3);
#pragma unroll
      for (int off = 1; off < 16; off <<= 1)
        rs += __shfl_xor(rs, off, 16);
      lrow[r] = lrow[r] * f + rs;
      mrow[r] = mn;
      fexp[r] = f;
      // stash P tile (C-layout row = r + 8*half, cols c*16 + l)
      __bf16* pr = pbuf + (r + 8 * hl) * 64;
      pr[l]      = f2bf(p0);
      pr[16 + l] = f2bf(p1);
      pr[32 + l] = f2bf(p2);
      pr[48 + l] = f2bf(p3);
    }
#pragma unroll
    for (int dt = 0; dt < 4; ++dt)
#pragma unroll
      for (int r = 0; r < 8; ++r)
        o[dt][r] *= fexp[r];

    // ---- reload P in A-fragment layout from LDS (same-wave, in-order DS) ----
    BF16Frag ap[2];
    {
      const __bf16* prow = pbuf + l * 64;
#pragma unroll
      for (int u = 0; u < 2; ++u) {
        ap[u].h[0] = *(const v8bf*)(prow + u * 32 + hl * 8);
        ap[u].h[1] = *(const v8bf*)(prow + u * 32 + 16 + hl * 8);
      }
    }

    // ---- O += P @ V : four 16x16 d-tiles x two 32-deep kv steps ----
#pragma unroll
    for (int dt = 0; dt < 4; ++dt) {
      const __bf16* vr = Vbase + (size_t)(dt * 16 + l) * Nkv + j0;
#pragma unroll
      for (int u = 0; u < 2; ++u) {
        BF16Frag bv;
        bv.h[0] = *(const v8bf*)(vr + u * 32 + hl * 8);
        bv.h[1] = *(const v8bf*)(vr + u * 32 + 16 + hl * 8);
        o[dt] = __builtin_amdgcn_wmma_f32_16x16x32_bf16(
            false, ap[u].v, false, bv.v, (short)0, o[dt], false, false);
      }
    }
  }

  // ---- normalize and store to AO [B*Nq, H*64] ----
  float inv[8];
#pragma unroll
  for (int r = 0; r < 8; ++r) inv[r] = 1.0f / lrow[r];
  const int b = bh / Hn, h = bh % Hn;
  const size_t rowStride = (size_t)Hn * 64;
#pragma unroll
  for (int dt = 0; dt < 4; ++dt)
#pragma unroll
    for (int r = 0; r < 8; ++r) {
      size_t orow = (size_t)b * Nq + q0 + hl * 8 + r;
      AO[orow * rowStride + h * 64 + dt * 16 + l] = f2bf(o[dt][r] * inv[r]);
    }
}

// ------------------------------- host side ---------------------------------

extern "C" void kernel_launch(void* const* d_in, const int* in_sizes, int n_in,
                              void* d_out, int out_size, void* d_ws, size_t ws_size,
                              hipStream_t stream)
{
  const int B = 4, Nq = 2048, Nkv = 2048, Cq = 1024, Ckv = 768, H = 16;
  (void)in_sizes; (void)n_in; (void)out_size; (void)ws_size;

  const float* query = (const float*)d_in[0];
  const float* keyv  = (const float*)d_in[1];
  const float* Wq = (const float*)d_in[2];
  const float* bq = (const float*)d_in[3];
  const float* Wk = (const float*)d_in[4];
  const float* bk = (const float*)d_in[5];
  const float* Wv = (const float*)d_in[6];
  const float* bv = (const float*)d_in[7];
  const float* Wo = (const float*)d_in[8];
  const float* bo = (const float*)d_in[9];

  char* ws = (char*)d_ws;
  size_t off = 0;
  auto take = [&](size_t bytes) -> void* {
    void* p = ws + off;
    off += (bytes + 255) & ~(size_t)255;
    return p;
  };

  __bf16* queryB = (__bf16*)take((size_t)B * Nq  * Cq  * 2);
  __bf16* kvB    = (__bf16*)take((size_t)B * Nkv * Ckv * 2);
  __bf16* WqT    = (__bf16*)take((size_t)Cq  * Cq * 2);
  __bf16* WkT    = (__bf16*)take((size_t)Ckv * Cq * 2);
  __bf16* WvT    = (__bf16*)take((size_t)Ckv * Cq * 2);
  __bf16* WoT    = (__bf16*)take((size_t)Cq  * Cq * 2);
  __bf16* Qh     = (__bf16*)take((size_t)B * Nq  * Cq * 2);
  __bf16* Kh     = (__bf16*)take((size_t)B * Nkv * Cq * 2);
  __bf16* Vt     = (__bf16*)take((size_t)B * Nkv * Cq * 2);
  __bf16* AO     = (__bf16*)take((size_t)B * Nq  * Cq * 2);

  // 1) activation conversions
  {
    int n0 = B * Nq * Cq;
    cvt_f32_bf16<<<(n0 + 255) / 256, 256, 0, stream>>>(query, queryB, n0);
    int n1 = B * Nkv * Ckv;
    cvt_f32_bf16<<<(n1 + 255) / 256, 256, 0, stream>>>(keyv, kvB, n1);
  }
  // 2) weight transpose+convert
  {
    int nqq = Cq * Cq, nkk = Ckv * Cq;
    xpose_f32_bf16<<<(nqq + 255) / 256, 256, 0, stream>>>(Wq, WqT, Cq,  Cq);
    xpose_f32_bf16<<<(nkk + 255) / 256, 256, 0, stream>>>(Wk, WkT, Ckv, Cq);
    xpose_f32_bf16<<<(nkk + 255) / 256, 256, 0, stream>>>(Wv, WvT, Ckv, Cq);
    xpose_f32_bf16<<<(nqq + 255) / 256, 256, 0, stream>>>(Wo, WoT, Cq,  Cq);
  }

  const int Mq  = B * Nq;   // 8192
  const int Mkv = B * Nkv;  // 8192
  dim3 blk(256);
  dim3 gq(Cq / 128, Mq / 128);    // 8 x 64
  dim3 gkv(Cq / 128, Mkv / 128);

  // 3) projections (Q, K head-major; V head-transposed)
  gemm_bf16_wmma<<<gq,  blk, 0, stream>>>(queryB, WqT, bq, Qh, Mq,  Cq, Cq,  1, Nq,  H);
  gemm_bf16_wmma<<<gkv, blk, 0, stream>>>(kvB,    WkT, bk, Kh, Mkv, Cq, Ckv, 1, Nkv, H);
  gemm_bf16_wmma<<<gkv, blk, 0, stream>>>(kvB,    WvT, bv, Vt, Mkv, Cq, Ckv, 3, Nkv, H);

  // 4) attention
  dim3 ga(Nq / 64, B * H);
  flash_attn_bf16<<<ga, dim3(128), 0, stream>>>(Qh, Kh, Vt, AO, Nq, Nkv, H);

  // 5) output projection -> fp32 d_out
  gemm_bf16_wmma<<<gq, blk, 0, stream>>>(AO, WoT, bo, (float*)d_out,
                                         Mq, Cq, Cq, 2, Nq, H);
}
